// FederatedAttention_39341900431459
// MI455X (gfx1250) — compile-verified, matrix operand order checked by reading, not run on hardware
//
#include <hip/hip_runtime.h>
#include <hip/hip_bf16.h>

typedef __attribute__((ext_vector_type(16))) __bf16 v16bf;
typedef __attribute__((ext_vector_type(8)))  float  v8f;

union FragB16 {
    v16bf v;
    uint4 q[2];
};

__device__ __forceinline__ unsigned short f2bf(float f) {
    unsigned int u = __float_as_uint(f);
    unsigned int r = u + 0x7FFFu + ((u >> 16) & 1u);   // round-to-nearest-even
    return (unsigned short)(r >> 16);
}

__device__ __forceinline__ v8f wmma_bf16(v16bf a, v16bf b, v8f c) {
    return __builtin_amdgcn_wmma_f32_16x16x32_bf16(
        /*neg_a=*/false, a, /*neg_b=*/false, b,
        /*c_mod=*/(short)0, c, /*reuse_a=*/false, /*reuse_b=*/false);
}

// 16x32 bf16 operand fragment from a row-major-along-K global array.
// lane 0-15 : row = row0+lane,    k = k0+{0..7} and k0+{16..23}
// lane 16-31: row = row0+lane-16, k = k0+{8..15} and k0+{24..31}
__device__ __forceinline__ v16bf load_frag(const unsigned short* __restrict__ base,
                                           int row0, int ld, int k0) {
    const int lane = threadIdx.x & 31;
    const int lr = lane & 15, half = lane >> 4;
    const unsigned short* p = base + (size_t)(row0 + lr) * ld + k0 + half * 8;
    FragB16 f;
    f.q[0] = *(const uint4*)(p);
    f.q[1] = *(const uint4*)(p + 16);
    return f.v;
}

// Fragment read from a 64-row x 512-col bf16 LDS tile stored with an XOR
// swizzle on 16B chunks: element chunk cc of row rr lives at chunk (cc ^ rr).
// Conflict-free ds_read_b128: 16 lanes hit 16 disjoint 4-bank groups.
__device__ __forceinline__ v16bf lds_frag_swz(const unsigned short* wlds,
                                              int row0, int k0) {
    const int lane = threadIdx.x & 31;
    const int lr = lane & 15, half = lane >> 4;
    const int rr = row0 + lr;              // 0..63
    const int c0 = (k0 >> 3) + half;       // 8-element chunk index
    FragB16 f;
    f.q[0] = *(const uint4*)(wlds + (((rr << 6) | ((c0)     ^ rr)) << 3));
    f.q[1] = *(const uint4*)(wlds + (((rr << 6) | ((c0 + 2) ^ rr)) << 3));
    return f.v;
}

// ---------------------------------------------------------------- convert ---
__global__ __launch_bounds__(256) void convert_bf16(const float* __restrict__ in,
                                                    unsigned short* __restrict__ out,
                                                    int n) {
    int i = (blockIdx.x * blockDim.x + threadIdx.x) * 4;
    if (i < n) {
        float4 f = *(const float4*)(in + i);
        unsigned int lo = (unsigned int)f2bf(f.x) | ((unsigned int)f2bf(f.y) << 16);
        unsigned int hi = (unsigned int)f2bf(f.z) | ((unsigned int)f2bf(f.w) << 16);
        *(uint2*)(out + i) = uint2{lo, hi};
    }
}

// --------------------------------------------------------------- QKV GEMM ---
// qkv[bn][d] = sum_c xb[bn][c] * wqkv[d][c].  Block: 128 rows x 64 cols.
// W tile (64x512 bf16 = 64KB) staged once per block in swizzled LDS; the
// fully-unrolled k-loop reads 4 B fragments per step into distinct registers
// (one LDS clause) then runs the 4 WMMAs; A fragments pipelined from global.
__global__ __launch_bounds__(256) void qkv_gemm(const unsigned short* __restrict__ xb,
                                                const unsigned short* __restrict__ wq,
                                                unsigned short* __restrict__ Q,
                                                unsigned short* __restrict__ K,
                                                unsigned short* __restrict__ Vt) {
    __shared__ __align__(16) unsigned short wlds[64 * 512];   // 64 KB
    const int wave = threadIdx.x >> 5;
    const int lane = threadIdx.x & 31;
    const int lr = lane & 15, half = lane >> 4;
    const int m0 = blockIdx.x * 128 + wave * 16;   // flat bn row
    const int nb = blockIdx.y * 64;                // output-feature base

    // Cooperative, coalesced stage of the 64x512 W tile with XOR chunk swizzle.
    {
        const unsigned short* wsrc = wq + (size_t)nb * 512;
        for (int e = threadIdx.x; e < 4096; e += 256) {        // 4096 16B chunks
            const int rr = e >> 6, cc = e & 63;
            uint4 d = *(const uint4*)(wsrc + ((rr << 9) | (cc << 3)));
            *(uint4*)(wlds + (((rr << 6) | (cc ^ rr)) << 3)) = d;
        }
    }
    __syncthreads();

    v8f acc[4];
#pragma unroll
    for (int t = 0; t < 4; ++t)
#pragma unroll
        for (int e = 0; e < 8; ++e) acc[t][e] = 0.0f;

    v16bf a_cur = load_frag(xb, m0, 512, 0);
#pragma unroll
    for (int k0 = 0; k0 < 512; k0 += 32) {        // fully unrolled: guard folds
        v16bf a_nxt = a_cur;
        if (k0 + 32 < 512) a_nxt = load_frag(xb, m0, 512, k0 + 32);  // pipeline
        v16bf bf[4];
#pragma unroll
        for (int t = 0; t < 4; ++t) bf[t] = lds_frag_swz(wlds, t * 16, k0);
#pragma unroll
        for (int t = 0; t < 4; ++t) acc[t] = wmma_bf16(a_cur, bf[t], acc[t]);
        a_cur = a_nxt;
    }

    const int bidx = m0 >> 12;       // batch
    const int n    = m0 & 4095;      // token
#pragma unroll
    for (int t = 0; t < 4; ++t) {
        const int dglob = nb + t * 16 + lr;     // segment uniform across the tile
        const int seg = dglob >> 9;             // 0=Q 1=K 2=V
        const int dm  = dglob & 511;
        const int h   = dm >> 6;
        const int dh  = dm & 63;
        const size_t bh = (size_t)bidx * 8 + h;
#pragma unroll
        for (int v = 0; v < 8; ++v) {
            const int row = n + v + (half << 3);
            const float val = acc[t][v];
            if (seg == 0)      Q[(bh * 4096 + row) * 64 + dh] = f2bf(val * 0.125f); // fold 1/sqrt(64)
            else if (seg == 1) K[(bh * 4096 + row) * 64 + dh] = f2bf(val);
            else               Vt[(bh * 64 + dh) * 4096 + row] = f2bf(val);
        }
    }
}

// ---------------------------------------------------------- flash attention --
// One wave owns 16 query rows; online softmax over j tiles of 32, unrolled by
// two with ping-pong K-fragment buffers (no register rotation): each K set is
// issued a full softmax+PV phase before its use.
#define LOG2E 1.4426950408889634f
__global__ __launch_bounds__(256) void flash_attn(const unsigned short* __restrict__ Q,
                                                  const unsigned short* __restrict__ K,
                                                  const unsigned short* __restrict__ Vt,
                                                  unsigned short* __restrict__ ao) {
    __shared__ __align__(16) unsigned short pshare[8][16 * 40];  // per-wave P tile
    const int wave = threadIdx.x >> 5;
    const int lane = threadIdx.x & 31;
    const int lr = lane & 15, half = lane >> 4;
    const int b = blockIdx.z, h = blockIdx.y;
    const size_t bh = (size_t)b * 8 + h;
    const int i0 = blockIdx.x * 128 + wave * 16;

    const unsigned short* Qb = Q + bh * 4096 * 64;
    const unsigned short* Kb = K + bh * 4096 * 64;
    const unsigned short* Vb = Vt + bh * 64 * 4096;
    unsigned short* pbuf = &pshare[wave][0];

    const v16bf qf0 = load_frag(Qb, i0, 64, 0);    // k = 0..31
    const v16bf qf1 = load_frag(Qb, i0, 64, 32);   // k = 32..63

    v8f o[4];
    float m[8], l[8];
#pragma unroll
    for (int t = 0; t < 4; ++t)
#pragma unroll
        for (int e = 0; e < 8; ++e) o[t][e] = 0.0f;
#pragma unroll
    for (int v = 0; v < 8; ++v) { m[v] = -3.0e38f; l[v] = 0.0f; }

    // Per-j-chunk body: S WMMAs from kf, online softmax, P->LDS->A-frag, PV.
    auto process = [&](int j0, const v16bf* kf) {
        // V fragments for this chunk: issue first, consumed after softmax.
        v16bf vf[4];
#pragma unroll
        for (int t = 0; t < 4; ++t) vf[t] = load_frag(Vb, t * 16, 4096, j0);
        if (j0 + 32 < 4096) {
            // Warm next chunk of V^T one phase early (global_prefetch_b8).
#pragma unroll
            for (int t = 0; t < 4; ++t)
                __builtin_prefetch(Vb + (size_t)(t * 16 + lr) * 4096 + j0 + 32 + half * 8, 0, 1);
        }

        v8f sA, sB;
#pragma unroll
        for (int e = 0; e < 8; ++e) { sA[e] = 0.0f; sB[e] = 0.0f; }
        sA = wmma_bf16(qf0, kf[0], sA);
        sA = wmma_bf16(qf1, kf[1], sA);
        sB = wmma_bf16(qf0, kf[2], sB);
        sB = wmma_bf16(qf1, kf[3], sB);

        // Online softmax: each row's 16 columns live in one 16-lane half.
        float alpha[8];
#pragma unroll
        for (int v = 0; v < 8; ++v) {
            float t = fmaxf(sA[v], sB[v]);
#pragma unroll
            for (int msk = 1; msk < 16; msk <<= 1) t = fmaxf(t, __shfl_xor(t, msk, 32));
            const float mn = fmaxf(m[v], t);
            alpha[v] = exp2f((m[v] - mn) * LOG2E);
            m[v] = mn;
            const float pA = exp2f((sA[v] - mn) * LOG2E);
            const float pB = exp2f((sB[v] - mn) * LOG2E);
            float rs = pA + pB;
#pragma unroll
            for (int msk = 1; msk < 16; msk <<= 1) rs += __shfl_xor(rs, msk, 32);
            l[v] = l[v] * alpha[v] + rs;
            // Spill P (bf16) to LDS in [i][j] order for A-operand re-read.
            const int row = v + (half << 3);
            pbuf[row * 40 + lr]      = f2bf(pA);
            pbuf[row * 40 + 16 + lr] = f2bf(pB);
        }
#pragma unroll
        for (int t = 0; t < 4; ++t)
#pragma unroll
            for (int v = 0; v < 8; ++v) o[t][v] *= alpha[v];

        // Re-read P as A fragment (same-wave LDS ops are in-order; 16B aligned).
        FragB16 pf;
        const unsigned short* pr = pbuf + lr * 40 + half * 8;
        pf.q[0] = *(const uint4*)(pr);
        pf.q[1] = *(const uint4*)(pr + 16);
#pragma unroll
        for (int t = 0; t < 4; ++t) o[t] = wmma_bf16(pf.v, vf[t], o[t]);
    };

    // Ping-pong K-fragment pipeline, two 32-wide j chunks per outer iteration.
    v16bf ka[4], kb[4];
    ka[0] = load_frag(Kb, 0,  64, 0);
    ka[1] = load_frag(Kb, 0,  64, 32);
    ka[2] = load_frag(Kb, 16, 64, 0);
    ka[3] = load_frag(Kb, 16, 64, 32);

    for (int j0 = 0; j0 < 4096; j0 += 64) {
        // Phase 0: preload kb for j0+32 (always in range), process j0 with ka.
        kb[0] = load_frag(Kb, j0 + 32, 64, 0);
        kb[1] = load_frag(Kb, j0 + 32, 64, 32);
        kb[2] = load_frag(Kb, j0 + 48, 64, 0);
        kb[3] = load_frag(Kb, j0 + 48, 64, 32);
        process(j0, ka);

        // Phase 1: preload ka for j0+64 (guarded), process j0+32 with kb.
        if (j0 + 64 < 4096) {
            ka[0] = load_frag(Kb, j0 + 64, 64, 0);
            ka[1] = load_frag(Kb, j0 + 64, 64, 32);
            ka[2] = load_frag(Kb, j0 + 80, 64, 0);
            ka[3] = load_frag(Kb, j0 + 80, 64, 32);
        }
        process(j0 + 32, kb);
    }

#pragma unroll
    for (int v = 0; v < 8; ++v) {
        const float inv = 1.0f / l[v];
#pragma unroll
        for (int t = 0; t < 4; ++t) o[t][v] *= inv;
    }
#pragma unroll
    for (int t = 0; t < 4; ++t)
#pragma unroll
        for (int v = 0; v < 8; ++v) {
            const int row = i0 + v + (half << 3);
            ao[((size_t)b * 4096 + row) * 512 + h * 64 + t * 16 + lr] = f2bf(o[t][v]);
        }
}

// --------------------------------------------- projection + residual + LN ---
// Block: 32 rows x full 512 cols. 8 waves = 2 row-tiles x 4 col-groups(128).
__global__ __launch_bounds__(256) void proj_ln(const unsigned short* __restrict__ ao,
                                               const unsigned short* __restrict__ wp,
                                               const float* __restrict__ x,
                                               const float* __restrict__ bproj,
                                               const float* __restrict__ gamma,
                                               const float* __restrict__ beta,
                                               float* __restrict__ out) {
    __shared__ float psum[32][4];
    __shared__ float psq[32][4];
    const int wave = threadIdx.x >> 5;
    const int lane = threadIdx.x & 31;
    const int lr = lane & 15, half = lane >> 4;
    const int mrow = wave & 1, cgrp = wave >> 1;
    const int m0 = blockIdx.x * 32 + mrow * 16;
    const int c0 = cgrp * 128;

    v8f acc[8];
#pragma unroll
    for (int t = 0; t < 8; ++t)
#pragma unroll
        for (int e = 0; e < 8; ++e) acc[t][e] = 0.0f;

    v16bf a_cur = load_frag(ao, m0, 512, 0);
#pragma unroll
    for (int k0 = 0; k0 < 512; k0 += 32) {        // fully unrolled: guard folds
        v16bf a_nxt = a_cur;
        if (k0 + 32 < 512) a_nxt = load_frag(ao, m0, 512, k0 + 32);  // pipeline
#pragma unroll
        for (int g = 0; g < 2; ++g) {          // two groups of 4 B fragments
            v16bf bf[4];
#pragma unroll
            for (int t = 0; t < 4; ++t)
                bf[t] = load_frag(wp, c0 + (g * 4 + t) * 16, 512, k0);
#pragma unroll
            for (int t = 0; t < 4; ++t)
                acc[g * 4 + t] = wmma_bf16(a_cur, bf[t], acc[g * 4 + t]);
        }
        a_cur = a_nxt;
    }

    float s1[8], s2[8];
#pragma unroll
    for (int v = 0; v < 8; ++v) { s1[v] = 0.0f; s2[v] = 0.0f; }
#pragma unroll
    for (int t = 0; t < 8; ++t) {
        const int col = c0 + t * 16 + lr;
        const float bp = bproj[col];
#pragma unroll
        for (int v = 0; v < 8; ++v) {
            const int row = m0 + v + (half << 3);
            const float y = acc[t][v] + bp + x[(size_t)row * 512 + col];
            acc[t][v] = y;                       // keep y for the LN pass
            s1[v] += y;
            s2[v] += y * y;
        }
    }
#pragma unroll
    for (int v = 0; v < 8; ++v)
#pragma unroll
        for (int msk = 1; msk < 16; msk <<= 1) {
            s1[v] += __shfl_xor(s1[v], msk, 32);
            s2[v] += __shfl_xor(s2[v], msk, 32);
        }
    if (lr == 0) {   // lanes 0 and 16 publish their half's 8 rows
#pragma unroll
        for (int v = 0; v < 8; ++v) {
            const int lrow = mrow * 16 + v + (half << 3);
            psum[lrow][cgrp] = s1[v];
            psq[lrow][cgrp]  = s2[v];
        }
    }
    __syncthreads();

    float mean[8], rstd[8];
#pragma unroll
    for (int v = 0; v < 8; ++v) {
        const int lrow = mrow * 16 + v + (half << 3);
        const float a1 = psum[lrow][0] + psum[lrow][1] + psum[lrow][2] + psum[lrow][3];
        const float a2 = psq[lrow][0] + psq[lrow][1] + psq[lrow][2] + psq[lrow][3];
        const float mu = a1 * (1.0f / 512.0f);
        const float var = a2 * (1.0f / 512.0f) - mu * mu;
        mean[v] = mu;
        rstd[v] = rsqrtf(var + 1e-5f);
    }
#pragma unroll
    for (int t = 0; t < 8; ++t) {
        const int col = c0 + t * 16 + lr;
        const float g = gamma[col], be = beta[col];
#pragma unroll
        for (int v = 0; v < 8; ++v) {
            const int row = m0 + v + (half << 3);
            out[(size_t)row * 512 + col] = (acc[t][v] - mean[v]) * rstd[v] * g + be;
        }
    }
}

// ---------------------------------------------------------------- launcher ---
extern "C" void kernel_launch(void* const* d_in, const int* in_sizes, int n_in,
                              void* d_out, int out_size, void* d_ws, size_t ws_size,
                              hipStream_t stream) {
    const float* x      = (const float*)d_in[0];   // [2,4096,512]
    const float* w_qkv  = (const float*)d_in[1];   // [1536,512]
    const float* w_proj = (const float*)d_in[2];   // [512,512]
    const float* b_proj = (const float*)d_in[3];   // [512]
    const float* gamma  = (const float*)d_in[4];   // [512]
    const float* beta   = (const float*)d_in[5];   // [512]
    float* out = (float*)d_out;

    char* ws = (char*)d_ws;
    unsigned short* xb     = (unsigned short*)(ws);                 // 8192x512   (8.0 MB)
    unsigned short* wqkvb  = (unsigned short*)(ws + 8388608);       // 1536x512   (1.5 MB)
    unsigned short* wprojb = (unsigned short*)(ws + 9961472);       // 512x512    (0.5 MB)
    unsigned short* Qws    = (unsigned short*)(ws + 10485760);      // [b,h,n,d]  (8.0 MB)
    unsigned short* Kws    = (unsigned short*)(ws + 18874368);      // [b,h,n,d]  (8.0 MB)
    unsigned short* Vtws   = (unsigned short*)(ws + 27262976);      // [b,h,d,n]  (8.0 MB)
    unsigned short* aows   = (unsigned short*)(ws + 35651584);      // [bn,512]   (8.0 MB)

    convert_bf16<<<4096, 256, 0, stream>>>(x, xb, 8192 * 512);
    convert_bf16<<<768,  256, 0, stream>>>(w_qkv, wqkvb, 1536 * 512);
    convert_bf16<<<256,  256, 0, stream>>>(w_proj, wprojb, 512 * 512);

    qkv_gemm<<<dim3(64, 24), 256, 0, stream>>>(xb, wqkvb, Qws, Kws, Vtws);

    flash_attn<<<dim3(32, 8, 2), 256, 0, stream>>>(Qws, Kws, Vtws, aows);

    proj_ln<<<256, 256, 0, stream>>>(aows, wprojb, x, b_proj, gamma, beta, out);
}
